// SEALGCN_53266184405212
// MI455X (gfx1250) — matrix operand order repaired and checked by reading, not compiled, not run on hardware
//
#include <hip/hip_runtime.h>
#include <hip/hip_bf16.h>
#include <math.h>

#define Nn 524288
#define Ee 4194304
#define Bb 4096
#define Hh 32
#define Kk 30

typedef __attribute__((ext_vector_type(16))) __bf16 v16bf;
typedef __attribute__((ext_vector_type(8)))  __bf16 v8bf;
typedef __attribute__((ext_vector_type(8)))  float  v8f;

__device__ __forceinline__ __bf16 f2bf(float f) {
  unsigned u = __builtin_bit_cast(unsigned, f);
  unsigned r = (u + 0x7FFFu + ((u >> 16) & 1u)) >> 16;
  unsigned short s = (unsigned short)r;
  return __builtin_bit_cast(__bf16, s);
}

__device__ __forceinline__ v8f wmma_bf16(v16bf a, v16bf b, v8f c) {
  // 8 args: (neg_a, A, neg_b, B, c_mod, C, reuse_a, reuse_b)
  return __builtin_amdgcn_wmma_f32_16x16x32_bf16(false, a, false, b, (short)0, c, false, false);
}

// Load the A operand: two contiguous 8-element bf16 halves -> 2x global_load_b128
__device__ __forceinline__ v16bf load_a16(const __bf16* p0, const __bf16* p1) {
  v8bf lo = *(const v8bf*)p0;
  v8bf hi = *(const v8bf*)p1;
  return __builtin_shufflevector(lo, hi, 0, 1, 2, 3, 4, 5, 6, 7, 8, 9, 10, 11, 12, 13, 14, 15);
}

// ---------------- degree / normalization ----------------
__global__ void k_deg_init(float* deg) {
  int i = blockIdx.x * blockDim.x + threadIdx.x;
  if (i < Nn) deg[i] = 1.0f;                       // self-loop
}
__global__ void k_deg_acc(const int* __restrict__ dst, float* deg) {
  int e = blockIdx.x * blockDim.x + threadIdx.x;
  if (e < Ee) atomicAdd(&deg[dst[e]], 1.0f);
}
__global__ void k_rsqrt(float* dinv) {
  int i = blockIdx.x * blockDim.x + threadIdx.x;
  if (i < Nn) dinv[i] = rsqrtf(dinv[i]);
}

// ---------------- embedding gather (f32 -> bf16) ----------------
__global__ void k_embed(const int* __restrict__ x_z, const float* __restrict__ emb,
                        __bf16* __restrict__ xb) {
  size_t i = (size_t)blockIdx.x * blockDim.x + threadIdx.x;   // N*32 threads
  int n = (int)(i >> 5), c = (int)(i & 31);
  xb[i] = f2bf(emb[(size_t)x_z[n] * Hh + c]);
}

// ---------------- weight conversion to bf16 ----------------
__global__ void k_cvtw(const float* __restrict__ Wc, const float* __restrict__ W1,
                       const float* __restrict__ W2,
                       __bf16* Wb, __bf16* W1b, __bf16* W2b) {
  int i = blockIdx.x * blockDim.x + threadIdx.x;
  if (i < 3072)                Wb[i]          = f2bf(Wc[i]);        // 3*32*32
  else if (i < 3072 + 30720)   W1b[i - 3072]  = f2bf(W1[i - 3072]); // 960*32
  else if (i < 3072 + 30720 + 512) W2b[i - 33792] = f2bf(W2[i - 33792]); // 32*16
}

// ---------------- GCN layer GEMM: h = x@Wc ; agg = h*self_norm + bc ----------------
// one wave -> 16 rows x 32 cols, K=32 -> one wmma per 16x16 tile.
// W staged in LDS pre-swizzled to the per-lane WMMA B layout: lane's 16 bf16 contiguous.
__global__ void k_gcn_gemm(const __bf16* __restrict__ xb, const __bf16* __restrict__ Wb,
                           const float* __restrict__ dinv, const float* __restrict__ bc,
                           float* __restrict__ h, float* __restrict__ agg) {
  __shared__ __attribute__((aligned(32))) __bf16 sW[1024]; // [tile(2)][lane(32)][tt(16)]
  int t = threadIdx.x;
  for (int i = t; i < 1024; i += blockDim.x) {
    int tile = i >> 9, ln = (i >> 4) & 31, tt = i & 15;
    int hf = ln >> 4, lcc = ln & 15;
    sW[i] = Wb[(hf * 8 + (tt & 7) + (tt >> 3) * 16) * 32 + tile * 16 + lcc];
  }
  __syncthreads();

  int wave = blockIdx.x * (blockDim.x >> 5) + (t >> 5);
  int lane = t & 31, half = lane >> 4;
  int m0 = wave * 16;

  // A tile 16x32 bf16 (lanes 0-15: K 0-7 / 16-23; lanes 16-31: K 8-15 / 24-31)
  const __bf16* ar = xb + (size_t)(m0 + (lane & 15)) * Hh;
  int kb0 = half ? 8 : 0, kb1 = half ? 24 : 16;
  v16bf a = load_a16(ar + kb0, ar + kb1);

  float sn[8];
#pragma unroll
  for (int r = 0; r < 8; ++r) { float d = dinv[m0 + half * 8 + r]; sn[r] = d * d; }

#pragma unroll
  for (int tile = 0; tile < 2; ++tile) {
    int n0 = tile * 16;
    v16bf b = *(const v16bf*)&sW[tile * 512 + lane * 16];   // 2x ds_load_b128
    v8f c = {};
    c = wmma_bf16(a, b, c);
    float bias = bc[n0 + (lane & 15)];
#pragma unroll
    for (int r = 0; r < 8; ++r) {
      int row = m0 + half * 8 + r;
      float v = c[r];
      h[(size_t)row * Hh + n0 + (lane & 15)]   = v;
      agg[(size_t)row * Hh + n0 + (lane & 15)] = v * sn[r] + bias;
    }
  }
}

// ---------------- edge scatter: agg[dst] += h[src] * dinv[src]*dinv[dst] ----------------
// one wave per edge (32 lanes = 32 channels); h/agg are L2-resident (64MB each < 192MB L2)
__global__ void k_scatter(const int* __restrict__ src, const int* __restrict__ dst,
                          const float* __restrict__ dinv, const float* __restrict__ h,
                          float* __restrict__ agg) {
  long long tid = (long long)blockIdx.x * blockDim.x + threadIdx.x;
  int e = (int)(tid >> 5);
  if (e >= Ee) return;
  int c = (int)(tid & 31);
  int s = src[e], d = dst[e];
  float w = dinv[s] * dinv[d];
  atomicAdd(&agg[(size_t)d * Hh + c], h[(size_t)s * Hh + c] * w);
}

// ---------------- BatchNorm(eval) + ReLU; write f32 and bf16 activations ----------------
__global__ void k_bn(const float* __restrict__ agg, const float* __restrict__ rm,
                     const float* __restrict__ rv, const float* __restrict__ g,
                     const float* __restrict__ bt, float* __restrict__ x,
                     __bf16* __restrict__ xb) {
  size_t i = (size_t)blockIdx.x * blockDim.x + threadIdx.x;  // N*32 threads
  int c = (int)(i & 31);
  float v = (agg[i] - rm[c]) * rsqrtf(rv[c] + 1e-5f) * g[c] + bt[c];
  v = fmaxf(v, 0.0f);
  x[i] = v;
  xb[i] = f2bf(v);
}

// ---------------- per-graph start offsets via binary search (batch is sorted) ----------------
__global__ void k_starts(const int* __restrict__ batch, int* __restrict__ starts) {
  int b = blockIdx.x * blockDim.x + threadIdx.x;
  if (b > Bb) return;
  int lo = 0, hi = Nn;
  while (lo < hi) { int mid = (lo + hi) >> 1; if (batch[mid] < b) lo = mid + 1; else hi = mid; }
  starts[b] = lo;
}

// ---------------- sort-pooling: top-K by last channel, stable (smallest index on ties) ----------------
__global__ void k_sortpool(const float* __restrict__ x, const int* __restrict__ starts,
                           __bf16* __restrict__ pooledb) {
  __shared__ float vals[1024];
  __shared__ float rv[256];
  __shared__ int   ri[256];
  __shared__ int   winner;
  int b = blockIdx.x, t = threadIdx.x;
  int n0 = starts[b];
  int cnt = starts[b + 1] - n0;
  if (cnt > 1024) cnt = 1024;   // avg count is 128; 1024 is >8 sigma headroom

  for (int j = t; j < 1024; j += 256)
    vals[j] = (j < cnt) ? x[(size_t)(n0 + j) * Hh + (Hh - 1)] : -3.4e38f;
  for (int j = t; j < Kk * Hh; j += 256)
    pooledb[(size_t)b * (Kk * Hh) + j] = f2bf(0.0f);   // zero-pad
  __syncthreads();

  int kmax = cnt < Kk ? cnt : Kk;
  for (int k = 0; k < kmax; ++k) {
    float bestv = -3.4e38f; int bestj = -1;
    for (int j = t; j < 1024; j += 256) {
      float v = vals[j];
      if (v > bestv) { bestv = v; bestj = j; }   // strided j ascending -> first hit is smallest j
    }
    rv[t] = bestv; ri[t] = bestj;
    __syncthreads();
    for (int s = 128; s > 0; s >>= 1) {
      if (t < s) {
        float va = rv[t], vb = rv[t + s];
        int ia = ri[t], ib = ri[t + s];
        if (ib >= 0 && (vb > va || (vb == va && (ia < 0 || ib < ia)))) { rv[t] = vb; ri[t] = ib; }
      }
      __syncthreads();
    }
    if (t == 0) { winner = ri[0]; vals[winner] = -3.4e38f; }
    __syncthreads();
    if (t < Hh)
      pooledb[(size_t)b * (Kk * Hh) + k * Hh + t] = f2bf(x[(size_t)(n0 + winner) * Hh + t]);
    __syncthreads();
  }
}

// ---------------- MLP1: [4096,960]@[960,32] + b1, relu -> bf16; W1 swizzled in LDS ----------------
__global__ void k_mlp1(const __bf16* __restrict__ pb, const __bf16* __restrict__ W1b,
                       const float* __restrict__ b1, __bf16* __restrict__ h1b) {
  __shared__ __attribute__((aligned(32))) __bf16 sW[30720]; // [kc(30)][tile(2)][lane(32)][tt(16)]
  int t = threadIdx.x;
  for (int i = t; i < 30720; i += blockDim.x) {
    int kc = i >> 10, rem = i & 1023;
    int tile = rem >> 9, ln = (rem >> 4) & 31, tt = rem & 15;
    int hf = ln >> 4, lcc = ln & 15;
    sW[i] = W1b[(kc * 32 + hf * 8 + (tt & 7) + (tt >> 3) * 16) * 32 + tile * 16 + lcc];
  }
  __syncthreads();

  int wave = blockIdx.x * (blockDim.x >> 5) + (t >> 5);   // 256 waves total
  int lane = t & 31, half = lane >> 4, lc = lane & 15;
  int m0 = wave * 16;
  const __bf16* ar = pb + (size_t)(m0 + lc) * 960;
  int kb0 = half ? 8 : 0, kb1 = half ? 24 : 16;

  v8f acc0 = {}, acc1 = {};
  for (int kc = 0; kc < 30; ++kc) {
    int kk = kc * 32;
    v16bf a  = load_a16(ar + kk + kb0, ar + kk + kb1);
    v16bf b0 = *(const v16bf*)&sW[kc * 1024 + lane * 16];
    v16bf b1v = *(const v16bf*)&sW[kc * 1024 + 512 + lane * 16];
    acc0 = wmma_bf16(a, b0, acc0);
    acc1 = wmma_bf16(a, b1v, acc1);
  }
#pragma unroll
  for (int r = 0; r < 8; ++r) {
    int row = m0 + half * 8 + r;
    h1b[(size_t)row * 32 + lc]      = f2bf(fmaxf(acc0[r] + b1[lc], 0.0f));
    h1b[(size_t)row * 32 + 16 + lc] = f2bf(fmaxf(acc1[r] + b1[16 + lc], 0.0f));
  }
}

// ---------------- MLP2: [4096,32]@[32,16] + b2, relu -> f32 (one wmma per 16 rows) ----------------
__global__ void k_mlp2(const __bf16* __restrict__ h1b, const __bf16* __restrict__ W2b,
                       const float* __restrict__ b2, float* __restrict__ h2) {
  __shared__ __attribute__((aligned(32))) __bf16 sW[512];  // [lane(32)][tt(16)]
  int t = threadIdx.x;
  for (int i = t; i < 512; i += blockDim.x) {
    int ln = (i >> 4) & 31, tt = i & 15;
    int hf = ln >> 4, lcc = ln & 15;
    sW[i] = W2b[(hf * 8 + (tt & 7) + (tt >> 3) * 16) * 16 + lcc];
  }
  __syncthreads();
  int wave = blockIdx.x * (blockDim.x >> 5) + (t >> 5);
  int lane = t & 31, half = lane >> 4, lc = lane & 15;
  int m0 = wave * 16;
  const __bf16* ar = h1b + (size_t)(m0 + lc) * 32;
  int kb0 = half ? 8 : 0, kb1 = half ? 24 : 16;
  v16bf a = load_a16(ar + kb0, ar + kb1);
  v16bf b = *(const v16bf*)&sW[lane * 16];
  v8f c = {};
  c = wmma_bf16(a, b, c);
#pragma unroll
  for (int r = 0; r < 8; ++r) {
    int row = m0 + half * 8 + r;
    h2[(size_t)row * 16 + lc] = fmaxf(c[r] + b2[lc], 0.0f);
  }
}

// ---------------- final: out[b] = h2[b,:] . W3 + b3 ----------------
__global__ void k_out(const float* __restrict__ h2, const float* __restrict__ W3,
                      const float* __restrict__ b3, float* __restrict__ out) {
  int b = blockIdx.x * blockDim.x + threadIdx.x;
  if (b >= Bb) return;
  float s = b3[0];
#pragma unroll
  for (int j = 0; j < 16; ++j) s += h2[(size_t)b * 16 + j] * W3[j];
  out[b] = s;
}

extern "C" void kernel_launch(void* const* d_in, const int* in_sizes, int n_in,
                              void* d_out, int out_size, void* d_ws, size_t ws_size,
                              hipStream_t stream) {
  (void)in_sizes; (void)n_in; (void)out_size; (void)ws_size;
  const int*   x_z   = (const int*)d_in[0];
  const int*   src   = (const int*)d_in[1];
  const int*   dst   = src + Ee;
  const int*   batch = (const int*)d_in[2];
  const float* emb   = (const float*)d_in[3];
  const float* Wc    = (const float*)d_in[4];
  const float* bc    = (const float*)d_in[5];
  const float* gamma = (const float*)d_in[6];
  const float* beta  = (const float*)d_in[7];
  const float* rmean = (const float*)d_in[8];
  const float* rvar  = (const float*)d_in[9];
  const float* W1    = (const float*)d_in[10];
  const float* b1    = (const float*)d_in[11];
  const float* W2    = (const float*)d_in[12];
  const float* b2    = (const float*)d_in[13];
  const float* W3    = (const float*)d_in[14];
  const float* b3    = (const float*)d_in[15];
  float* out = (float*)d_out;

  char* ws = (char*)d_ws;
  size_t off = 0;
  auto alloc = [&](size_t bytes) { void* p = ws + off; off = (off + bytes + 255) & ~(size_t)255; return p; };

  float*  dinv    = (float*)alloc((size_t)Nn * 4);
  __bf16* xb      = (__bf16*)alloc((size_t)Nn * Hh * 2);
  float*  x       = (float*)alloc((size_t)Nn * Hh * 4);
  float*  h       = (float*)alloc((size_t)Nn * Hh * 4);
  float*  agg     = (float*)alloc((size_t)Nn * Hh * 4);
  __bf16* Wb      = (__bf16*)alloc(3 * 1024 * 2);
  __bf16* W1b     = (__bf16*)alloc(30720 * 2);
  __bf16* W2b     = (__bf16*)alloc(512 * 2);
  int*    starts  = (int*)alloc((Bb + 1) * 4);
  __bf16* pooledb = (__bf16*)alloc((size_t)Bb * Kk * Hh * 2);
  __bf16* h1b     = (__bf16*)alloc((size_t)Bb * 32 * 2);
  float*  h2      = (float*)alloc((size_t)Bb * 16 * 4);

  k_deg_init<<<Nn / 256, 256, 0, stream>>>(dinv);
  k_deg_acc<<<Ee / 256, 256, 0, stream>>>(dst, dinv);
  k_rsqrt<<<Nn / 256, 256, 0, stream>>>(dinv);
  k_embed<<<(Nn * Hh) / 256, 256, 0, stream>>>(x_z, emb, xb);
  k_cvtw<<<(34304 + 255) / 256, 256, 0, stream>>>(Wc, W1, W2, Wb, W1b, W2b);

  for (int i = 0; i < 3; ++i) {
    k_gcn_gemm<<<Nn / 16 / 8, 256, 0, stream>>>(xb, Wb + i * 1024, dinv, bc + i * Hh, h, agg);
    k_scatter<<<((size_t)Ee * 32) / 256, 256, 0, stream>>>(src, dst, dinv, h, agg);
    k_bn<<<(Nn * Hh) / 256, 256, 0, stream>>>(agg, rmean + i * Hh, rvar + i * Hh,
                                              gamma + i * Hh, beta + i * Hh, x, xb);
  }

  k_starts<<<(Bb + 1 + 255) / 256, 256, 0, stream>>>(batch, starts);
  k_sortpool<<<Bb, 256, 0, stream>>>(x, starts, pooledb);
  k_mlp1<<<64, 128, 0, stream>>>(pooledb, W1b, b1, h1b);   // 256 waves of 16 rows
  k_mlp2<<<64, 128, 0, stream>>>(h1b, W2b, b2, h2);
  k_out<<<Bb / 256, 256, 0, stream>>>(h2, W3, b3, out);
}
